// LucyRNN_48515950576464
// MI455X (gfx1250) — compile-verified, matrix operand order checked by reading, not run on hardware
//
#include <hip/hip_runtime.h>
#include <hip/hip_bf16.h>
#include <stdint.h>

// Problem sizes (fixed by the reference)
#define Bc   8
#define Tc   2048
#define Hc   512
#define Lc   2
#define Vc   4096
#define BTc  (Bc * Tc)          // 16384 rows
#define EPSc 1e-5f

// scan chunking for the time recurrences
#define NCH  32
#define CLEN (Tc / NCH)         // 64

typedef unsigned short u16;
typedef __attribute__((ext_vector_type(16))) __bf16 v16bf;
typedef __attribute__((ext_vector_type(8)))  float  v8f;

union Frag {
    v16bf v;
    unsigned int u[8];
    uint4 q[2];
};

__device__ __forceinline__ u16 f2bf(float f) {
    unsigned int x = __float_as_uint(f);
    x += 0x7FFFu + ((x >> 16) & 1u);   // round-to-nearest-even
    return (u16)(x >> 16);
}

// gfx1250 async copy: global -> LDS, 16 bytes per lane, tracked by ASYNCcnt
__device__ __forceinline__ void async_copy_b128(unsigned lds_off, const void* gaddr) {
    asm volatile("global_load_async_to_lds_b128 %0, %1, off"
                 :: "v"(lds_off), "v"(gaddr)
                 : "memory");
}
__device__ __forceinline__ void wait_async0() {
    asm volatile("s_wait_asynccnt 0x0" ::: "memory");
}
// allow 2 in flight (the next slab's two copies) -> oldest slab is complete
__device__ __forceinline__ void wait_async2() {
    asm volatile("s_wait_asynccnt 0x2" ::: "memory");
}

// ---------------------------------------------------------------------------
// C[M,N] (f32) = A[M,K] (bf16 row-major) * Bt[N,K] (bf16, W transposed) + bias
// block = 128 threads (4 waves). Block tile 128(M) x 64(N); wave tile 32 x 64.
// B slab (64 x 32 bf16 = 4KB) double-buffered in LDS via async-to-LDS with
// pipelined ASYNCcnt waits; 8 v_wmma per k-step per wave.
// ---------------------------------------------------------------------------
__global__ void __launch_bounds__(128)
gemm_bf16_wmma(const u16* __restrict__ A, const u16* __restrict__ Bt,
               const float* __restrict__ bias, float* __restrict__ C,
               int M, int N, int K)
{
    __shared__ __align__(16) u16 sh[2][64 * 32];  // two 4KB slabs [n' (64)][k' (32)]

    const int tid  = threadIdx.x;
    const int lane = tid & 31;
    const int wave = tid >> 5;
    const int n0   = blockIdx.x * 64;
    const int mw   = blockIdx.y * 128 + wave * 32;
    const int mr   = lane & 15;
    const int half = lane >> 4;

    const u16* arow0 = A + (size_t)(mw + mr) * K;
    const u16* arow1 = arow0 + (size_t)16 * K;

    // async-copy assignment: 256 x 16B chunks per slab, 2 per thread
    const unsigned shb[2] = { (unsigned)(unsigned long long)&sh[0][0],
                              (unsigned)(unsigned long long)&sh[1][0] };
    const int c0 = tid;
    const int c1 = tid + 128;
    const u16* g0p = Bt + (size_t)(n0 + (c0 >> 2)) * K + (c0 & 3) * 8;
    const u16* g1p = Bt + (size_t)(n0 + (c1 >> 2)) * K + (c1 & 3) * 8;
    const unsigned lo0 = (unsigned)c0 * 16u;
    const unsigned lo1 = (unsigned)c1 * 16u;

    v8f acc[2][4];
#pragma unroll
    for (int mi = 0; mi < 2; ++mi)
#pragma unroll
        for (int j = 0; j < 4; ++j) acc[mi][j] = (v8f){};

    // prologue: stage k-step 0 into buffer 0
    async_copy_b128(shb[0] + lo0, g0p);
    async_copy_b128(shb[0] + lo1, g1p);

    const int nsteps = K >> 5;                    // K/32
    for (int ks = 0; ks < nsteps; ++ks) {
        const int k0  = ks << 5;
        const int cur = ks & 1;

        if (ks + 1 < nsteps) {
            // kick off next slab into the other buffer, then wait only for
            // the current slab's (oldest two) copies to land
            async_copy_b128(shb[cur ^ 1] + lo0, g0p + k0 + 32);
            async_copy_b128(shb[cur ^ 1] + lo1, g1p + k0 + 32);
            wait_async2();
        } else {
            wait_async0();
        }
        __syncthreads();   // publish current slab to all 4 waves

        // A fragments (direct global; rows are private to this wave)
        Frag a0, a1;
        a0.q[0] = *(const uint4*)(arow0 + k0 + 8 * half);
        a0.q[1] = *(const uint4*)(arow0 + k0 + 16 + 8 * half);
        a1.q[0] = *(const uint4*)(arow1 + k0 + 8 * half);
        a1.q[1] = *(const uint4*)(arow1 + k0 + 16 + 8 * half);
        if (k0 + 128 < K) {
            __builtin_prefetch(arow0 + k0 + 128, 0, 1);   // global_prefetch_b8
            __builtin_prefetch(arow1 + k0 + 128, 0, 1);
        }

        // B fragments from LDS
        const u16* shcur = &sh[cur][0];
        Frag bf[4];
#pragma unroll
        for (int j = 0; j < 4; ++j) {
            const u16* r = shcur + (j * 16 + mr) * 32;
            bf[j].q[0] = *(const uint4*)(r + 8 * half);
            bf[j].q[1] = *(const uint4*)(r + 16 + 8 * half);
        }

#pragma unroll
        for (int j = 0; j < 4; ++j) {
            acc[0][j] = __builtin_amdgcn_wmma_f32_16x16x32_bf16(
                false, a0.v, false, bf[j].v, (short)0, acc[0][j], false, false);
            acc[1][j] = __builtin_amdgcn_wmma_f32_16x16x32_bf16(
                false, a1.v, false, bf[j].v, (short)0, acc[1][j], false, false);
        }
        __syncthreads();   // all waves done reading before this buffer is reused
    }

    // C/D layout: n = lane%16, m = vgpr + 8*(lane/16)
#pragma unroll
    for (int j = 0; j < 4; ++j) {
        const int ncol = n0 + j * 16 + mr;
        const float bv = bias ? bias[ncol] : 0.0f;
#pragma unroll
        for (int mi = 0; mi < 2; ++mi) {
#pragma unroll
            for (int r = 0; r < 8; ++r) {
                const int m = mw + mi * 16 + r + 8 * half;
                C[(size_t)m * N + ncol] = acc[mi][j][r] + bv;
            }
        }
    }
}

// ---------------------------------------------------------------------------
// Row LayerNorm over H=512 + optional activation. 128 threads per row.
// mode 0: y (+ bf16 copy if outb) | mode 1: sigmoid(y) | mode 2: tanh(y)
// ---------------------------------------------------------------------------
__global__ void __launch_bounds__(128)
ln_act(const float* __restrict__ in, const float* __restrict__ g,
       const float* __restrict__ b, float* __restrict__ outf,
       u16* __restrict__ outb, int mode)
{
    const int row  = blockIdx.x;
    const int tid  = threadIdx.x;
    const int lane = tid & 31;
    const int wave = tid >> 5;
    const float* x = in + (size_t)row * Hc;

    float v[4];
    float s = 0.0f, s2 = 0.0f;
#pragma unroll
    for (int i = 0; i < 4; ++i) {
        v[i] = x[tid + i * 128];
        s  += v[i];
        s2 += v[i] * v[i];
    }
#pragma unroll
    for (int o = 16; o > 0; o >>= 1) {
        s  += __shfl_xor(s,  o, 32);
        s2 += __shfl_xor(s2, o, 32);
    }
    __shared__ float sm[8];
    if (lane == 0) { sm[wave] = s; sm[4 + wave] = s2; }
    __syncthreads();
    s  = sm[0] + sm[1] + sm[2] + sm[3];
    s2 = sm[4] + sm[5] + sm[6] + sm[7];

    const float mean = s * (1.0f / Hc);
    const float var  = s2 * (1.0f / Hc) - mean * mean;
    const float rstd = rsqrtf(var + EPSc);

    float* yo = outf + (size_t)row * Hc;
#pragma unroll
    for (int i = 0; i < 4; ++i) {
        const int col = tid + i * 128;
        float y = (v[i] - mean) * rstd * g[col] + b[col];
        if (mode == 1)      y = 1.0f / (1.0f + __expf(-y));
        else if (mode == 2) y = tanhf(y);
        yo[col] = y;
        if (outb) outb[(size_t)row * Hc + col] = f2bf(y);
    }
}

// ---------------------------------------------------------------------------
// Decayed suffix sum, chunked two-pass (32 chunks x 64 steps).
// Phase 1: part[(b*NCH+ch)*H + h] = sum_{t in chunk} kv[t]*e^{p*t}
// ---------------------------------------------------------------------------
__global__ void decay_partial(const float* __restrict__ kk, const float* __restrict__ vv,
                              const float* __restrict__ pd, float* __restrict__ part)
{
    const int idx = blockIdx.x * blockDim.x + threadIdx.x;   // B*NCH*H
    if (idx >= Bc * NCH * Hc) return;
    const int h  = idx % Hc;
    const int ch = (idx / Hc) % NCH;
    const int b  = idx / (Hc * NCH);

    const float p = pd[h];
    float s = 0.0f;
    const int tbase = ch * CLEN;
    for (int i = 0; i < CLEN; ++i) {
        const int t = tbase + i;
        const size_t off = ((size_t)b * Tc + t) * Hc + h;
        s += kk[off] * vv[off] * __expf(p * (float)t);
    }
    part[idx] = s;
}

// Phase 2: s_all[t] = tail(chunks > ch) + in-chunk suffix; write bf16(u + decay*s_all + kv)
__global__ void decay_apply(const float* __restrict__ kk, const float* __restrict__ vv,
                            const float* __restrict__ u, const float* __restrict__ pd,
                            const float* __restrict__ part, u16* __restrict__ usb)
{
    const int idx = blockIdx.x * blockDim.x + threadIdx.x;
    if (idx >= Bc * NCH * Hc) return;
    const int h  = idx % Hc;
    const int ch = (idx / Hc) % NCH;
    const int b  = idx / (Hc * NCH);

    const float p     = pd[h];
    const float decay = __expf(p);
    float s_all = 0.0f;
    for (int cc = ch + 1; cc < NCH; ++cc)
        s_all += part[((size_t)b * NCH + cc) * Hc + h];

    for (int i = CLEN - 1; i >= 0; --i) {
        const int t = ch * CLEN + i;
        const size_t off = ((size_t)b * Tc + t) * Hc + h;
        const float kv = kk[off] * vv[off];
        s_all += kv * __expf(p * (float)t);
        usb[off] = f2bf(u[off] + decay * s_all + kv);
    }
}

// ---------------------------------------------------------------------------
// Gated scan h_t = z_t*h_{t-1} + (1-z_t)*c_t as a chunked linear-recurrence scan.
// Phase 1: per chunk compose (A,B): h_out = A*h_in + B
// ---------------------------------------------------------------------------
__global__ void gru_partial(const float* __restrict__ z, const float* __restrict__ cc,
                            float* __restrict__ pA, float* __restrict__ pB)
{
    const int idx = blockIdx.x * blockDim.x + threadIdx.x;
    if (idx >= Bc * NCH * Hc) return;
    const int h  = idx % Hc;
    const int ch = (idx / Hc) % NCH;
    const int b  = idx / (Hc * NCH);

    float Aacc = 1.0f, Bacc = 0.0f;
    const int tbase = ch * CLEN;
    for (int i = 0; i < CLEN; ++i) {
        const size_t off = ((size_t)b * Tc + tbase + i) * Hc + h;
        const float zt = z[off];
        const float wt = (1.0f - zt) * cc[off];
        Aacc = zt * Aacc;
        Bacc = zt * Bacc + wt;
    }
    pA[idx] = Aacc;
    pB[idx] = Bacc;
}

// Phase 2: fold preceding chunk pairs to get h_in, then replay chunk, write bf16
__global__ void gru_apply(const float* __restrict__ z, const float* __restrict__ cc,
                          const float* __restrict__ pA, const float* __restrict__ pB,
                          u16* __restrict__ hb)
{
    const int idx = blockIdx.x * blockDim.x + threadIdx.x;
    if (idx >= Bc * NCH * Hc) return;
    const int h  = idx % Hc;
    const int ch = (idx / Hc) % NCH;
    const int b  = idx / (Hc * NCH);

    float hcur = 0.0f;
    for (int p = 0; p < ch; ++p) {
        const size_t pidx = ((size_t)b * NCH + p) * Hc + h;
        hcur = pA[pidx] * hcur + pB[pidx];
    }
    const int tbase = ch * CLEN;
    for (int i = 0; i < CLEN; ++i) {
        const size_t off = ((size_t)b * Tc + tbase + i) * Hc + h;
        const float zt = z[off];
        hcur = zt * hcur + (1.0f - zt) * cc[off];
        hb[off] = f2bf(hcur);
    }
}

// ---------------------------------------------------------------------------
// Conversions
// ---------------------------------------------------------------------------
__global__ void to_bf16(const float* __restrict__ in, u16* __restrict__ out, size_t n)
{
    const size_t i = (size_t)blockIdx.x * blockDim.x + threadIdx.x;
    if (i < n) out[i] = f2bf(in[i]);
}

// in: [K,N] f32 row-major  ->  out: [N,K] bf16 (transposed)
__global__ void transpose_bf16(const float* __restrict__ in, u16* __restrict__ out,
                               int K, int N)
{
    const size_t i = (size_t)blockIdx.x * blockDim.x + threadIdx.x;
    if (i >= (size_t)K * N) return;
    const int k = (int)(i / N);
    const int n = (int)(i - (size_t)k * N);
    out[(size_t)n * K + k] = f2bf(in[i]);
}

// ---------------------------------------------------------------------------
extern "C" void kernel_launch(void* const* d_in, const int* in_sizes, int n_in,
                              void* d_out, int out_size, void* d_ws, size_t ws_size,
                              hipStream_t stream)
{
    (void)in_sizes; (void)n_in; (void)out_size; (void)ws_size;

    const float* x    = (const float*)d_in[0];
    const float* Wi   = (const float*)d_in[1];
    const float* bi   = (const float*)d_in[2];
    const float* g_in = (const float*)d_in[3];
    const float* b_in = (const float*)d_in[4];
    const float* Wz   = (const float*)d_in[5];
    const float* bz   = (const float*)d_in[6];
    const float* g_z  = (const float*)d_in[7];
    const float* b_z  = (const float*)d_in[8];
    const float* Wk   = (const float*)d_in[9];
    const float* bk   = (const float*)d_in[10];
    const float* Wv   = (const float*)d_in[11];
    const float* bv   = (const float*)d_in[12];
    const float* Wh   = (const float*)d_in[13];
    const float* bh   = (const float*)d_in[14];
    const float* g_h  = (const float*)d_in[15];
    const float* b_h  = (const float*)d_in[16];
    const float* pdc  = (const float*)d_in[17];
    const float* Wo   = (const float*)d_in[18];
    const float* bo   = (const float*)d_in[19];
    float* out = (float*)d_out;

    // Workspace carve-up
    char* ws = (char*)d_ws;
    size_t off = 0;
    auto alloc = [&](size_t bytes) -> void* {
        void* p = ws + off;
        off = (off + bytes + 255) & ~(size_t)255;
        return p;
    };

    const size_t nBTH = (size_t)BTc * Hc;
    const size_t nPart = (size_t)Bc * NCH * Hc;
    u16* inpb = (u16*)alloc(nBTH * 2);               // layer input, bf16
    u16* ub   = (u16*)alloc(nBTH * 2);               // u, bf16
    u16* usb  = (u16*)alloc(nBTH * 2);               // u+s, bf16
    u16* wt[Lc][5];
    for (int l = 0; l < Lc; ++l)
        for (int w = 0; w < 5; ++w)
            wt[l][w] = (u16*)alloc((size_t)Hc * Hc * 2);
    u16* wo_t = (u16*)alloc((size_t)Vc * Hc * 2);
    float* g0 = (float*)alloc(nBTH * 4);             // gemm scratch / k
    float* uB = (float*)alloc(nBTH * 4);             // u (f32)
    float* zB = (float*)alloc(nBTH * 4);             // z
    float* vB = (float*)alloc(nBTH * 4);             // v
    float* cB = (float*)alloc(nBTH * 4);             // c
    float* part = (float*)alloc(nPart * 4);          // chunk partials (decay)
    float* pA   = (float*)alloc(nPart * 4);          // chunk compose A (scan)
    float* pB_  = (float*)alloc(nPart * 4);          // chunk compose B (scan)

    // --- precision prep ---
    to_bf16<<<(unsigned)((nBTH + 255) / 256), 256, 0, stream>>>(x, inpb, nBTH);
    const float* Wsrc[5] = {Wi, Wz, Wk, Wv, Wh};
    const unsigned hhBlocks = (unsigned)(((size_t)Hc * Hc + 255) / 256);
    for (int l = 0; l < Lc; ++l)
        for (int w = 0; w < 5; ++w)
            transpose_bf16<<<hhBlocks, 256, 0, stream>>>(
                Wsrc[w] + (size_t)l * Hc * Hc, wt[l][w], Hc, Hc);
    transpose_bf16<<<(unsigned)(((size_t)Hc * Vc + 255) / 256), 256, 0, stream>>>(
        Wo, wo_t, Hc, Vc);

    const dim3 gblk(128);
    const dim3 ggridH(Hc / 64, BTc / 128);           // (8, 128)
    const dim3 ggridV(Vc / 64, BTc / 128);           // (64, 128)
    const unsigned scanBlocks = (unsigned)((nPart + 255) / 256);   // 512

    for (int l = 0; l < Lc; ++l) {
        // u = LN(inp @ Wi + bi)
        gemm_bf16_wmma<<<ggridH, gblk, 0, stream>>>(inpb, wt[l][0], bi + l * Hc, g0, BTc, Hc, Hc);
        ln_act<<<BTc, 128, 0, stream>>>(g0, g_in + l * Hc, b_in + l * Hc, uB, ub, 0);
        // z = sigmoid(LN(u @ Wz + bz))
        gemm_bf16_wmma<<<ggridH, gblk, 0, stream>>>(ub, wt[l][1], bz + l * Hc, g0, BTc, Hc, Hc);
        ln_act<<<BTc, 128, 0, stream>>>(g0, g_z + l * Hc, b_z + l * Hc, zB, nullptr, 1);
        // k, v
        gemm_bf16_wmma<<<ggridH, gblk, 0, stream>>>(ub, wt[l][2], bk + l * Hc, g0, BTc, Hc, Hc);
        gemm_bf16_wmma<<<ggridH, gblk, 0, stream>>>(ub, wt[l][3], bv + l * Hc, vB, BTc, Hc, Hc);
        // s (decayed suffix sum), chunk-parallel, fused u+s -> bf16
        decay_partial<<<scanBlocks, 256, 0, stream>>>(g0, vB, pdc + l * Hc, part);
        decay_apply<<<scanBlocks, 256, 0, stream>>>(g0, vB, uB, pdc + l * Hc, part, usb);
        // c = tanh(LN((u+s) @ Wh + bh))
        gemm_bf16_wmma<<<ggridH, gblk, 0, stream>>>(usb, wt[l][4], bh + l * Hc, g0, BTc, Hc, Hc);
        ln_act<<<BTc, 128, 0, stream>>>(g0, g_h + l * Hc, b_h + l * Hc, cB, nullptr, 2);
        // h scan (chunk-parallel linear recurrence) -> next layer input (bf16)
        gru_partial<<<scanBlocks, 256, 0, stream>>>(zB, cB, pA, pB_);
        gru_apply<<<scanBlocks, 256, 0, stream>>>(zB, cB, pA, pB_, inpb);
    }

    // out = h @ Wo + bo
    gemm_bf16_wmma<<<ggridV, gblk, 0, stream>>>(inpb, wo_t, bo, out, BTc, Vc, Hc);
}
